// ADConv_12086037971532
// MI455X (gfx1250) — compile-verified, matrix operand order checked by reading, not run on hardware
//
#include <hip/hip_runtime.h>
#include <hip/hip_bf16.h>
#include <math.h>

// ---------------------------------------------------------------------------
// Problem constants (from reference)
// ---------------------------------------------------------------------------
#define NN    8
#define CC    32
#define HH    64
#define WW    64
#define KS    7
#define NB    6
#define TB    18          // total bases
#define BCDIM 108         // NB*TB
#define INTER 64
#define OUTC  64
#define PIX   (HH*WW)     // 4096
#define NPIX  (NN*PIX)    // 32768

#define XPW   70          // xpad width  (halo 3 each side)
#define HPW   66          // hpad width  (halo 1 each side)

typedef __attribute__((ext_vector_type(16))) _Float16 v16h;
typedef __attribute__((ext_vector_type(8)))  _Float16 v8h;
typedef __attribute__((ext_vector_type(8)))  float    v8f;

union F16x16 { v16h v; v8h p[2]; _Float16 h[16]; };
union F32x8  { v8f  v; float f[8]; };

// 16-bit A/B fragment layout (CDNA5 ISA 7.12.2): per lane-half, elements
// 0..7 hold K = 8*half + 0..7 and elements 8..15 hold K = 16 + 8*half + 0..7,
// i.e. two contiguous 8-element K-runs -> two aligned 16B loads per fragment.

// ---------------------------------------------------------------------------
// K0a: pack weights to f16 with tap-major/channel-minor K order (k = r*CIN+c)
//      conv2 padded to 112 output rows with zeros
// ---------------------------------------------------------------------------
__global__ __launch_bounds__(256)
void pack_weights(const float* __restrict__ w1, const float* __restrict__ w2,
                  const float* __restrict__ w3,
                  _Float16* __restrict__ w1h, _Float16* __restrict__ w2h,
                  _Float16* __restrict__ w3h) {
    int i = blockIdx.x * 256 + threadIdx.x;
    if (i < 64 * 288) {                       // k2 = r*32+c  <-  src c*9+r
        int co = i / 288, k2 = i - co * 288;
        int r = k2 >> 5, c = k2 & 31;
        w1h[i] = (_Float16)w1[co * 288 + c * 9 + r];
    }
    if (i < 112 * 576) {                      // k2 = r*64+c  <-  src c*9+r
        int co = i / 576, k2 = i - co * 576;
        int r = k2 >> 6, c = k2 & 63;
        w2h[i] = (co < BCDIM) ? (_Float16)w2[co * 576 + c * 9 + r] : (_Float16)0.0f;
    }
    if (i < 64 * 192) w3h[i] = (_Float16)w3[i];   // K order already c*6+m
}

// ---------------------------------------------------------------------------
// K0b: x (NCHW f32) -> xpad (n,70,70,32) f16 channel-last, zero halo of 3
// ---------------------------------------------------------------------------
__global__ __launch_bounds__(256)
void pack_x(const float* __restrict__ x, _Float16* __restrict__ xpad) {
    int idx = blockIdx.x * 256 + threadIdx.x;
    if (idx >= NN * XPW * XPW * CC) return;
    int c = idx & 31;
    int t = idx >> 5;
    int ww = t % XPW; t /= XPW;
    int hh = t % XPW;
    int n  = t / XPW;
    float v = 0.0f;
    if (hh >= 3 && hh < 67 && ww >= 3 && ww < 67)
        v = x[(((size_t)n * CC + c) * 64 + (hh - 3)) * 64 + (ww - 3)];
    xpad[idx] = (_Float16)v;
}

// ---------------------------------------------------------------------------
// K1/K4: implicit-GEMM 3x3 conv + bias via v_wmma_f32_16x16x32_f16
//   A source: channel-last zero-halo f16 tensor; every 8-element fragment run
//   is one aligned v8h (16B) load -> global_load_b128.
//   block = 128 threads (4 waves); block = one (n,h) row of 64 pixels.
// ---------------------------------------------------------------------------
template <int CIN, int CTILES, int COUT_BIAS, int COUT_STORE, int PADW, int PADOFF>
__global__ __launch_bounds__(128)
void conv3x3_bias_wmma(const _Float16* __restrict__ in, const _Float16* __restrict__ wgt,
                       const float* __restrict__ bias, float* __restrict__ out) {
    constexpr int K   = CIN * 9;
    constexpr int KCH = K / 32;

    const int n    = blockIdx.x >> 6;
    const int h    = blockIdx.x & 63;
    const int wv   = threadIdx.x >> 5;
    const int lane = threadIdx.x & 31;
    const int half = lane >> 4;
    const int lcol = lane & 15;
    const int wpix = wv * 16 + lcol;   // A-row pixel for this lane
    const int w0   = wv * 16;

    // hint the weight panel into cache (global_prefetch_b8)
    __builtin_prefetch(&wgt[(size_t)lane * K], 0, 1);

    v8f acc[CTILES];
#pragma unroll
    for (int nt = 0; nt < CTILES; ++nt) acc[nt] = {};

#pragma unroll 1
    for (int kc = 0; kc < KCH; ++kc) {
        // A fragment: two aligned 16B runs of 8 consecutive channels
        F16x16 Af;
#pragma unroll
        for (int run = 0; run < 2; ++run) {
            int k0  = kc * 32 + (run ? 16 : 0) + 8 * half;
            int r   = k0 / CIN;            // tap index 0..8 (uniform over run)
            int c0  = k0 - r * CIN;        // first channel of the run
            int row = h + r / 3 + PADOFF;
            int col = wpix + r % 3 + PADOFF;
            Af.p[run] = *(const v8h*)&in[((size_t)(n * PADW + row) * PADW + col) * CIN + c0];
        }
#pragma unroll
        for (int nt = 0; nt < CTILES; ++nt) {
            F16x16 Bf;
            const int co = nt * 16 + lcol;
#pragma unroll
            for (int run = 0; run < 2; ++run) {
                int k0 = kc * 32 + (run ? 16 : 0) + 8 * half;
                Bf.p[run] = *(const v8h*)&wgt[(size_t)co * K + k0];
            }
            acc[nt] = __builtin_amdgcn_wmma_f32_16x16x32_f16(
                false, Af.v, false, Bf.v, (short)0, acc[nt], false, false);
        }
    }

    // D layout: lane<16 -> M=vgpr, lane>=16 -> M=vgpr+8 ; N=lane&15
#pragma unroll
    for (int nt = 0; nt < CTILES; ++nt) {
        F32x8 D; D.v = acc[nt];
        const int co = nt * 16 + lcol;
        float bv = (co < COUT_BIAS) ? bias[co] : 0.0f;
#pragma unroll
        for (int rI = 0; rI < 8; ++rI) {
            int pix = w0 + rI + 8 * half;
            out[(((size_t)n * COUT_STORE + co) * 64 + h) * 64 + pix] = D.f[rI] + bv;
        }
    }
}

// ---------------------------------------------------------------------------
// K2/K5: deterministic per-channel BN stats -> fused scale/shift
// ---------------------------------------------------------------------------
__global__ __launch_bounds__(256)
void bn_stats(const float* __restrict__ y, const float* __restrict__ g,
              const float* __restrict__ b, float* __restrict__ scale,
              float* __restrict__ shift, int cstore) {
    __shared__ float s1[256], s2[256];
    const int c = blockIdx.x, t = threadIdx.x;
    float a1 = 0.0f, a2 = 0.0f;
    for (int i = t; i < NN * PIX; i += 256) {
        int n  = i >> 12, hw = i & 4095;
        float v = y[(((size_t)n * cstore + c) << 12) + hw];
        a1 += v; a2 += v * v;
    }
    s1[t] = a1; s2[t] = a2;
    __syncthreads();
    for (int s = 128; s > 0; s >>= 1) {
        if (t < s) { s1[t] += s1[t + s]; s2[t] += s2[t + s]; }
        __syncthreads();
    }
    if (t == 0) {
        const float inv = 1.0f / (float)(NN * PIX);
        float mean = s1[0] * inv;
        float var  = s2[0] * inv - mean * mean;
        float sc   = g[c] * rsqrtf(var + 1e-5f);
        scale[c] = sc;
        shift[c] = b[c] - mean * sc;
    }
}

// K3: BN+tanh on conv1 output -> hpad (n,66,66,64) f16 channel-last, halo 1
__global__ __launch_bounds__(256)
void bn1_apply(const float* __restrict__ y, const float* __restrict__ sc,
               const float* __restrict__ sh, _Float16* __restrict__ hpad) {
    int idx = blockIdx.x * 256 + threadIdx.x;
    if (idx >= NN * HPW * HPW * INTER) return;
    int c = idx & 63;
    int t = idx >> 6;
    int ww = t % HPW; t /= HPW;
    int hh = t % HPW;
    int n  = t / HPW;
    float v = 0.0f;
    if (hh >= 1 && hh < 65 && ww >= 1 && ww < 65) {
        float yv = y[(((size_t)n * INTER + c) * 64 + (hh - 1)) * 64 + (ww - 1)];
        v = tanhf(fmaf(yv, sc[c], sh[c]));
    }
    hpad[idx] = (_Float16)v;
}

// K6: BN+tanh on conv2 output, repack pixel-major: coefp[p][j], j = m*18+t
__global__ __launch_bounds__(256)
void bn2_apply_pack(const float* __restrict__ y2, const float* __restrict__ sc,
                    const float* __restrict__ sh, _Float16* __restrict__ coefp) {
    int idx = blockIdx.x * 256 + threadIdx.x;
    if (idx >= NN * BCDIM * PIX) return;
    int hw = idx & 4095;
    int t  = idx >> 12;
    int j  = t % BCDIM;
    int n  = t / BCDIM;
    float v  = y2[(((size_t)n * 112 + j) << 12) + hw];
    float cv = tanhf(fmaf(v, sc[j], sh[j]));
    coefp[((size_t)((n << 12) + hw)) * BCDIM + j] = (_Float16)cv;
}

// ---------------------------------------------------------------------------
// K7: fused per-pixel stage + WMMA output projection.
//   bases_out[c,m] = sum_t coef[m,t] * (sum_k patch[c,k]*bases[t,k])  (VALU)
//   patches come from xpad (halo 3 -> no bounds checks); staged to LDS f16,
//   then out[p,o] = bases_out[p,:192] . out_w[o,:192] via WMMA.
// ---------------------------------------------------------------------------
__global__ __launch_bounds__(64)
void final_fused(const _Float16* __restrict__ xpad, const float* __restrict__ bases,
                 const _Float16* __restrict__ coefp, const _Float16* __restrict__ w3h,
                 const float* __restrict__ outb, float* __restrict__ out) {
    __shared__ float basesS[TB * KS * KS];                  // 18*49 f32
    __shared__ __align__(16) _Float16 boS[64 * 192];        // bases_out, f16

    const int n = blockIdx.x >> 6, h = blockIdx.x & 63;
    const int tid = threadIdx.x;

    for (int i = tid; i < TB * KS * KS; i += 64) basesS[i] = bases[i];
    __syncthreads();

    const int w = tid;
    const size_t p = ((size_t)n << 12) + h * 64 + w;

    float cf[BCDIM];
#pragma unroll
    for (int j = 0; j < BCDIM; ++j) cf[j] = (float)coefp[p * BCDIM + j];

#pragma unroll 1
    for (int c = 0; c < CC; ++c) {
        float pk[KS * KS];
#pragma unroll
        for (int r = 0; r < KS * KS; ++r) {
            int row = h + r / 7, col = w + r % 7;   // xpad halo 3 absorbs pad
            pk[r] = (float)xpad[((size_t)(n * XPW + row) * XPW + col) * CC + c];
        }
        float g[TB];
#pragma unroll
        for (int t = 0; t < TB; ++t) {
            float s = 0.0f;
#pragma unroll
            for (int k = 0; k < KS * KS; ++k) s = fmaf(pk[k], basesS[t * 49 + k], s);
            g[t] = s;
        }
#pragma unroll
        for (int m = 0; m < NB; ++m) {
            float s = 0.0f;
#pragma unroll
            for (int t = 0; t < TB; ++t) s = fmaf(cf[m * TB + t], g[t], s);
            boS[w * 192 + c * NB + m] = (_Float16)s;
        }
    }
    __syncthreads();

    const int wv = tid >> 5, lane = tid & 31, half = lane >> 4, lcol = lane & 15;
#pragma unroll
    for (int mi = 0; mi < 2; ++mi) {
        const int mt = wv * 2 + mi;  // 16-pixel M-tile
#pragma unroll
        for (int nt = 0; nt < 4; ++nt) {
            v8f acc = {};
#pragma unroll
            for (int kc = 0; kc < 6; ++kc) {  // K = 192 = 6*32
                F16x16 Af, Bf;
#pragma unroll
                for (int run = 0; run < 2; ++run) {
                    int k0 = kc * 32 + (run ? 16 : 0) + 8 * half;
                    Af.p[run] = *(const v8h*)&boS[(mt * 16 + lcol) * 192 + k0];
                    Bf.p[run] = *(const v8h*)&w3h[(size_t)(nt * 16 + lcol) * 192 + k0];
                }
                acc = __builtin_amdgcn_wmma_f32_16x16x32_f16(
                    false, Af.v, false, Bf.v, (short)0, acc, false, false);
            }
            F32x8 D; D.v = acc;
            const int o  = nt * 16 + lcol;
            const float bo = outb[o];
#pragma unroll
            for (int rI = 0; rI < 8; ++rI) {
                int pix = mt * 16 + rI + 8 * half;
                out[(((size_t)n * OUTC + o) * 64 + h) * 64 + pix] = D.f[rI] + bo;
            }
        }
    }
}

// ---------------------------------------------------------------------------
// Workspace layout (bytes); all offsets 256-aligned
// ---------------------------------------------------------------------------
#define OFF_W1H    ((size_t)0)                         //  64*288 f16 = 36864
#define OFF_W2H    (OFF_W1H + 36864)                   // 112*576 f16 = 129024
#define OFF_W3H    (OFF_W2H + 129024)                  //  64*192 f16 = 24576
#define OFF_SC1    (OFF_W3H + 24576)
#define OFF_SH1    (OFF_SC1 + 256)
#define OFF_SC2    (OFF_SH1 + 256)
#define OFF_SH2    (OFF_SC2 + 512)
#define OFF_XPAD   (OFF_SH2 + 512)                     // 8*70*70*32 f16 = 2508800
#define OFF_Y1     (OFF_XPAD + 2508800)                // 8*64*4096 f32 = 8388608
#define OFF_HPAD   (OFF_Y1 + 8388608)                  // 8*66*66*64 f16 = 4460544
#define OFF_Y2     (OFF_HPAD + 4460544)                // 8*112*4096 f32 = 14680064
#define OFF_COEF   (OFF_Y2 + 14680064)                 // 32768*108 f16 = 7077888

extern "C" void kernel_launch(void* const* d_in, const int* in_sizes, int n_in,
                              void* d_out, int out_size, void* d_ws, size_t ws_size,
                              hipStream_t stream) {
    const float* x       = (const float*)d_in[0];
    const float* conv1_w = (const float*)d_in[1];
    const float* conv1_b = (const float*)d_in[2];
    const float* bn1_g   = (const float*)d_in[3];
    const float* bn1_b   = (const float*)d_in[4];
    const float* conv2_w = (const float*)d_in[5];
    const float* conv2_b = (const float*)d_in[6];
    const float* bn2_g   = (const float*)d_in[7];
    const float* bn2_b   = (const float*)d_in[8];
    const float* bases   = (const float*)d_in[9];
    const float* out_w   = (const float*)d_in[10];
    const float* out_b   = (const float*)d_in[11];

    char* ws = (char*)d_ws;
    _Float16* w1h   = (_Float16*)(ws + OFF_W1H);
    _Float16* w2h   = (_Float16*)(ws + OFF_W2H);
    _Float16* w3h   = (_Float16*)(ws + OFF_W3H);
    float*    sc1   = (float*)(ws + OFF_SC1);
    float*    sh1   = (float*)(ws + OFF_SH1);
    float*    sc2   = (float*)(ws + OFF_SC2);
    float*    sh2   = (float*)(ws + OFF_SH2);
    _Float16* xpad  = (_Float16*)(ws + OFF_XPAD);
    float*    y1    = (float*)(ws + OFF_Y1);
    _Float16* hpad  = (_Float16*)(ws + OFF_HPAD);
    float*    y2    = (float*)(ws + OFF_Y2);
    _Float16* coefp = (_Float16*)(ws + OFF_COEF);
    float*    out   = (float*)d_out;

    pack_weights<<<(112 * 576 + 255) / 256, 256, 0, stream>>>(
        conv1_w, conv2_w, out_w, w1h, w2h, w3h);

    pack_x<<<(NN * XPW * XPW * CC + 255) / 256, 256, 0, stream>>>(x, xpad);

    // conv1: reads xpad (halo 3, use inner offset +2 for 3x3)
    conv3x3_bias_wmma<32, 4, 64, 64, XPW, 2><<<NN * HH, 128, 0, stream>>>(
        xpad, w1h, conv1_b, y1);

    bn_stats<<<INTER, 256, 0, stream>>>(y1, bn1_g, bn1_b, sc1, sh1, INTER);

    bn1_apply<<<(NN * HPW * HPW * INTER + 255) / 256, 256, 0, stream>>>(
        y1, sc1, sh1, hpad);

    // conv2: reads hpad (halo 1, offset 0)
    conv3x3_bias_wmma<64, 7, 108, 112, HPW, 0><<<NN * HH, 128, 0, stream>>>(
        hpad, w2h, conv2_b, y2);

    bn_stats<<<BCDIM, 256, 0, stream>>>(y2, bn2_g, bn2_b, sc2, sh2, 112);

    bn2_apply_pack<<<(NN * BCDIM * PIX + 255) / 256, 256, 0, stream>>>(
        y2, sc2, sh2, coefp);

    final_fused<<<NN * HH, 64, 0, stream>>>(xpad, bases, coefp, w3h, out_b, out);
}
